// TorchAttention_87222195847864
// MI455X (gfx1250) — matrix operand order repaired and checked by reading, not compiled, not run on hardware
//
#include <hip/hip_runtime.h>

// Problem constants (from reference): B=2, S=2048, D=1024, H=16, hd=64
#define DM    1024
#define HEADS 16
#define HD    64
#define SEQ   2048
#define BATCH 2
#define MROWS (BATCH * SEQ) // 4096

typedef __attribute__((ext_vector_type(16))) __bf16 v16bf;
typedef __attribute__((ext_vector_type(8)))  float  v8f;
typedef __attribute__((ext_vector_type(4)))  unsigned int u32x4;

// Load 16 bf16 (elements 0..7 from p0, 8..15 from p1) as two 16B vector loads.
__device__ __forceinline__ v16bf load_frag16(const __bf16* p0, const __bf16* p1) {
  union { v16bf v; u32x4 u[2]; } f;
  f.u[0] = *reinterpret_cast<const u32x4*>(p0);
  f.u[1] = *reinterpret_cast<const u32x4*>(p1);
  return f.v;
}

__device__ __forceinline__ v8f wmma_bf16(v16bf a, v16bf b, v8f c) {
  // (neg_a, A, neg_b, B, c_mod, C, reuse_a, reuse_b)
  return __builtin_amdgcn_wmma_f32_16x16x32_bf16(false, a, false, b, (short)0, c,
                                                 false, false);
}

// A fragment (16x32 bf16): lane m=l%16 holds K runs [k0+h*8, +8) and [k0+16+h*8, +8)
__device__ __forceinline__ v16bf ldA(const __bf16* row, int k0, int half) {
  return load_frag16(row + k0 + half * 8, row + k0 + 16 + half * 8);
}
// B fragment (32x16 bf16): lane n=l%16 holds K runs [k0+h*8, +8) and [k0+16+h*8, +8)
__device__ __forceinline__ v16bf ldB(const __bf16* row, int k0, int half) {
  return load_frag16(row + k0 + half * 8, row + k0 + 16 + half * 8);
}

// ---------------------------------------------------------------------------
// fp32 -> bf16 conversion (4 elements / thread, vectorized load)
// ---------------------------------------------------------------------------
__global__ void cvt_f32_bf16(const float* __restrict__ in, __bf16* __restrict__ out,
                             int n4) {
  int i = blockIdx.x * blockDim.x + threadIdx.x;
  if (i < n4) {
    float4 v = reinterpret_cast<const float4*>(in)[i];
    out[4 * i + 0] = (__bf16)v.x;
    out[4 * i + 1] = (__bf16)v.y;
    out[4 * i + 2] = (__bf16)v.z;
    out[4 * i + 3] = (__bf16)v.w;
  }
}

// ---------------------------------------------------------------------------
// QKV projection: y = x @ W^T.  One wave -> 32 rows x 64 cols (one head slice).
// Software-pipelined: next k-step's fragments are in flight during 8 WMMAs.
// mode 0=Q (pre-scaled by 1/8) ,1=K -> [bh,s,64]; 2=V -> transposed [bh,64,s].
// ---------------------------------------------------------------------------
__global__ void qkv_proj(const __bf16* __restrict__ x, const __bf16* __restrict__ W,
                         __bf16* __restrict__ out, int mode) {
  const int wave = (blockIdx.x * blockDim.x + threadIdx.x) >> 5;
  const int lane = threadIdx.x & 31;
  const int half = lane >> 4, lm = lane & 15;
  const int mTile = wave >> 4;       // 0..127  (32-row tiles)
  const int h     = wave & 15;       // head
  const int m0    = mTile * 32;

  const __bf16* arow0 = x + (size_t)(m0 + lm) * DM;
  const __bf16* arow1 = arow0 + (size_t)16 * DM;
  const __bf16* wrow  = W + (size_t)(h * HD + lm) * DM; // + nf*16*DM per n-subtile

  v8f zero = {0.f, 0.f, 0.f, 0.f, 0.f, 0.f, 0.f, 0.f};
  v8f acc0[4] = {zero, zero, zero, zero};
  v8f acc1[4] = {zero, zero, zero, zero};

  // pipeline preload (k0 = 0)
  v16bf a0 = ldA(arow0, 0, half);
  v16bf a1 = ldA(arow1, 0, half);
  v16bf b[4];
#pragma unroll
  for (int nf = 0; nf < 4; ++nf) b[nf] = ldB(wrow + (size_t)nf * 16 * DM, 0, half);

  for (int k0 = 0; k0 < DM; k0 += 32) {
    const int kn = (k0 + 32) & (DM - 1); // wraps to 0 on last iter (values unused)
    v16bf a0n = ldA(arow0, kn, half);
    v16bf a1n = ldA(arow1, kn, half);
    v16bf bn[4];
#pragma unroll
    for (int nf = 0; nf < 4; ++nf)
      bn[nf] = ldB(wrow + (size_t)nf * 16 * DM, kn, half);

#pragma unroll
    for (int nf = 0; nf < 4; ++nf) {
      acc0[nf] = wmma_bf16(a0, b[nf], acc0[nf]);
      acc1[nf] = wmma_bf16(a1, b[nf], acc1[nf]);
    }
    a0 = a0n; a1 = a1n;
#pragma unroll
    for (int nf = 0; nf < 4; ++nf) b[nf] = bn[nf];
  }

  // Fold softmax scale 1/sqrt(64) into Q (power of two -> exact in bf16).
  const float sc = (mode == 0) ? 0.125f : 1.0f;

  const int b_   = m0 >> 11;                 // / SEQ
  const size_t bh = (size_t)b_ * HEADS + h;
#pragma unroll
  for (int sub = 0; sub < 2; ++sub) {
    v8f* acc = sub ? acc1 : acc0;
    const int sbase = (m0 & (SEQ - 1)) + sub * 16 + half * 8;
#pragma unroll
    for (int nf = 0; nf < 4; ++nf) {
#pragma unroll
      for (int r = 0; r < 8; ++r) {
        float v = acc[nf][r] * sc;
        if (mode < 2)
          out[(bh * SEQ + sbase + r) * HD + nf * 16 + lm] = (__bf16)v;
        else
          out[(bh * HD + nf * 16 + lm) * SEQ + sbase + r] = (__bf16)v;
      }
    }
  }
}

// ---------------------------------------------------------------------------
// Flash attention: one wave per 32-row Q tile, KV streamed in steps of 32.
// Q,K: [bh, s, 64] bf16 (Q pre-scaled by 1/8).  Vt: [bh, 64, s].  ctx: [4096,1024].
// K fragments double-buffered across iterations; V fragments issued before the
// score WMMAs so they fly during matrix + softmax work.
// ---------------------------------------------------------------------------
__global__ void attn_fwd(const __bf16* __restrict__ Q, const __bf16* __restrict__ K,
                         const __bf16* __restrict__ Vt, __bf16* __restrict__ ctx) {
  __shared__ __align__(16) __bf16 pbuf[4][2][16][32]; // per-wave P tiles

  const int wid  = threadIdx.x >> 5;
  const int lane = threadIdx.x & 31;
  const int half = lane >> 4, lm = lane & 15;
  const int gw = blockIdx.x * (blockDim.x >> 5) + wid; // 0..2047
  const int bh = gw >> 6;                              // 0..31
  const int qt = gw & 63;
  const int q0 = qt * 32;

  const __bf16* qbase = Q  + (size_t)bh * SEQ * HD;
  const __bf16* kbase = K  + (size_t)bh * SEQ * HD;
  const __bf16* vbase = Vt + (size_t)bh * HD * SEQ;

  // Q A-fragments: 2 row-subtiles x 2 k-steps, kept in registers.
  v16bf qa[2][2];
#pragma unroll
  for (int a = 0; a < 2; ++a) {
    const __bf16* qrow = qbase + (size_t)(q0 + a * 16 + lm) * HD;
    qa[a][0] = ldA(qrow, 0, half);
    qa[a][1] = ldA(qrow, 32, half);
  }

  v8f zero = {0.f, 0.f, 0.f, 0.f, 0.f, 0.f, 0.f, 0.f};
  v8f o[2][4];
  float mrow[2][8], lrow[2][8];
#pragma unroll
  for (int a = 0; a < 2; ++a) {
#pragma unroll
    for (int nf = 0; nf < 4; ++nf) o[a][nf] = zero;
#pragma unroll
    for (int r = 0; r < 8; ++r) { mrow[a][r] = -1e30f; lrow[a][r] = 0.f; }
  }

  // bk[ct*2+ks]: K B-fragment, column tile ct (j..j+16), k-step ks (d-range)
  v16bf bk[4];
#pragma unroll
  for (int ct = 0; ct < 2; ++ct) {
    const __bf16* kr = kbase + (size_t)(ct * 16 + lm) * HD;
    bk[ct * 2 + 0] = ldB(kr, 0, half);
    bk[ct * 2 + 1] = ldB(kr, 32, half);
  }

  for (int j0 = 0; j0 < SEQ; j0 += 32) {
    const int jn = (j0 + 32) & (SEQ - 1); // wraps on last iter (values unused)

    // issue next iteration's K fragments now (consumed next iter)
    v16bf bkn[4];
#pragma unroll
    for (int ct = 0; ct < 2; ++ct) {
      const __bf16* kr = kbase + (size_t)(jn + ct * 16 + lm) * HD;
      bkn[ct * 2 + 0] = ldB(kr, 0, half);
      bkn[ct * 2 + 1] = ldB(kr, 32, half);
    }
    // issue V fragments for this iteration (consumed after softmax)
    v16bf vb[4];
#pragma unroll
    for (int nf = 0; nf < 4; ++nf) {
      const __bf16* vr = vbase + (size_t)(nf * 16 + lm) * SEQ + j0;
      vb[nf] = ldB(vr, 0, half);
    }
    // prefetch j0+64 K/V lines into cache (global_prefetch_b8)
    {
      const int jp = (j0 + 64) & (SEQ - 1);
      __builtin_prefetch(kbase + (size_t)(jp + lm) * HD, 0, 3);
      __builtin_prefetch(vbase + (size_t)(lm * 2) * SEQ + jp, 0, 3);
    }

    // --- scores: 2 row-subtiles x 2 column tiles (Q pre-scaled by 1/8)
    v8f s[2][2];
#pragma unroll
    for (int a = 0; a < 2; ++a) {
#pragma unroll
      for (int ct = 0; ct < 2; ++ct) {
        v8f t = zero;
        t = wmma_bf16(qa[a][0], bk[ct * 2 + 0], t);
        t = wmma_bf16(qa[a][1], bk[ct * 2 + 1], t);
        s[a][ct] = t;
      }
    }

    // --- online softmax. D-layout: lane owns col n=lm of rows half*8 + r.
#pragma unroll
    for (int a = 0; a < 2; ++a) {
      float cv[8];
#pragma unroll
      for (int r = 0; r < 8; ++r) {
        float a0 = s[a][0][r];
        float a1 = s[a][1][r];
        float t = fmaxf(a0, a1);
        t = fmaxf(t, __shfl_xor(t, 1));
        t = fmaxf(t, __shfl_xor(t, 2));
        t = fmaxf(t, __shfl_xor(t, 4));
        t = fmaxf(t, __shfl_xor(t, 8));
        float mn = fmaxf(mrow[a][r], t);
        float p0 = __expf(a0 - mn);
        float p1 = __expf(a1 - mn);
        float c  = __expf(mrow[a][r] - mn);
        mrow[a][r] = mn;
        float rs = p0 + p1;
        rs += __shfl_xor(rs, 1);
        rs += __shfl_xor(rs, 2);
        rs += __shfl_xor(rs, 4);
        rs += __shfl_xor(rs, 8);
        lrow[a][r] = lrow[a][r] * c + rs;
        cv[r] = c;
        pbuf[wid][a][half * 8 + r][lm]      = (__bf16)p0;
        pbuf[wid][a][half * 8 + r][16 + lm] = (__bf16)p1;
      }
#pragma unroll
      for (int nf = 0; nf < 4; ++nf)
#pragma unroll
        for (int r = 0; r < 8; ++r) o[a][nf][r] *= cv[r];
    }

    __builtin_amdgcn_wave_barrier(); // order LDS writes before cross-lane reads

    v16bf pa[2];
#pragma unroll
    for (int a = 0; a < 2; ++a) {
      const __bf16* prow = &pbuf[wid][a][lm][0];
      pa[a] = load_frag16(prow + half * 8, prow + 16 + half * 8);
    }

    __builtin_amdgcn_wave_barrier(); // keep next-iter writes after these reads

    // --- O += P @ V
#pragma unroll
    for (int a = 0; a < 2; ++a)
#pragma unroll
      for (int nf = 0; nf < 4; ++nf)
        o[a][nf] = wmma_bf16(pa[a], vb[nf], o[a][nf]);

#pragma unroll
    for (int i = 0; i < 4; ++i) bk[i] = bkn[i];
  }

  // finalize + store ctx (merge heads): row = b*SEQ + s, col = h*64 + d
  const int b = bh >> 4, h = bh & 15;
#pragma unroll
  for (int a = 0; a < 2; ++a) {
    const int sbase = q0 + a * 16 + half * 8;
#pragma unroll
    for (int nf = 0; nf < 4; ++nf)
#pragma unroll
      for (int r = 0; r < 8; ++r)
        ctx[(size_t)(b * SEQ + sbase + r) * DM + h * HD + nf * 16 + lm] =
            (__bf16)(o[a][nf][r] / lrow[a][r]);
  }
}

// ---------------------------------------------------------------------------
// Output projection: out = ctx @ Wo^T, fp32 output.  32x64 tile per wave,
// software-pipelined like qkv_proj.
// ---------------------------------------------------------------------------
__global__ void o_proj(const __bf16* __restrict__ ctx, const __bf16* __restrict__ W,
                       float* __restrict__ out) {
  const int wave = (blockIdx.x * blockDim.x + threadIdx.x) >> 5;
  const int lane = threadIdx.x & 31;
  const int half = lane >> 4, lm = lane & 15;
  const int mTile = wave >> 4;   // 0..127
  const int nb    = wave & 15;
  const int m0 = mTile * 32, n0 = nb * 64;

  const __bf16* arow0 = ctx + (size_t)(m0 + lm) * DM;
  const __bf16* arow1 = arow0 + (size_t)16 * DM;
  const __bf16* wrow  = W + (size_t)(n0 + lm) * DM;

  v8f zero = {0.f, 0.f, 0.f, 0.f, 0.f, 0.f, 0.f, 0.f};
  v8f acc0[4] = {zero, zero, zero, zero};
  v8f acc1[4] = {zero, zero, zero, zero};

  v16bf a0 = ldA(arow0, 0, half);
  v16bf a1 = ldA(arow1, 0, half);
  v16bf b[4];
#pragma unroll
  for (int nf = 0; nf < 4; ++nf) b[nf] = ldB(wrow + (size_t)nf * 16 * DM, 0, half);

  for (int k0 = 0; k0 < DM; k0 += 32) {
    const int kn = (k0 + 32) & (DM - 1);
    v16bf a0n = ldA(arow0, kn, half);
    v16bf a1n = ldA(arow1, kn, half);
    v16bf bn[4];
#pragma unroll
    for (int nf = 0; nf < 4; ++nf)
      bn[nf] = ldB(wrow + (size_t)nf * 16 * DM, kn, half);

#pragma unroll
    for (int nf = 0; nf < 4; ++nf) {
      acc0[nf] = wmma_bf16(a0, b[nf], acc0[nf]);
      acc1[nf] = wmma_bf16(a1, b[nf], acc1[nf]);
    }
    a0 = a0n; a1 = a1n;
#pragma unroll
    for (int nf = 0; nf < 4; ++nf) b[nf] = bn[nf];
  }

#pragma unroll
  for (int sub = 0; sub < 2; ++sub) {
    v8f* acc = sub ? acc1 : acc0;
    const int mb = m0 + sub * 16 + half * 8;
#pragma unroll
    for (int nf = 0; nf < 4; ++nf)
#pragma unroll
      for (int r = 0; r < 8; ++r)
        out[(size_t)(mb + r) * DM + n0 + nf * 16 + lm] = acc[nf][r];
  }
}

// ---------------------------------------------------------------------------
extern "C" void kernel_launch(void* const* d_in, const int* in_sizes, int n_in,
                              void* d_out, int out_size, void* d_ws, size_t ws_size,
                              hipStream_t stream) {
  const float* x  = (const float*)d_in[0];
  const float* Wq = (const float*)d_in[1];
  const float* Wk = (const float*)d_in[2];
  const float* Wv = (const float*)d_in[3];
  const float* Wo = (const float*)d_in[4];
  float* out = (float*)d_out;

  const size_t MB = 1u << 20;
  char* ws = (char*)d_ws;
  __bf16* xb   = (__bf16*)(ws + 0 * MB);   // 8 MB:  x      [4096,1024]
  __bf16* wqb  = (__bf16*)(ws + 8 * MB);   // 2 MB each
  __bf16* wkb  = (__bf16*)(ws + 10 * MB);
  __bf16* wvb  = (__bf16*)(ws + 12 * MB);
  __bf16* wob  = (__bf16*)(ws + 14 * MB);
  __bf16* qb   = (__bf16*)(ws + 16 * MB);  // 8 MB:  Q   [bh,s,64] (pre-scaled)
  __bf16* kb   = (__bf16*)(ws + 24 * MB);  // 8 MB:  K   [bh,s,64]
  __bf16* vtb  = (__bf16*)(ws + 32 * MB);  // 8 MB:  V^T [bh,64,s]
  __bf16* ctxb = (__bf16*)(ws + 40 * MB);  // 8 MB:  ctx [4096,1024]

  // 1) convert to bf16
  cvt_f32_bf16<<<(MROWS * DM / 4 + 255) / 256, 256, 0, stream>>>(x, xb,
                                                                 MROWS * DM / 4);
  cvt_f32_bf16<<<(DM * DM / 4 + 255) / 256, 256, 0, stream>>>(Wq, wqb, DM * DM / 4);
  cvt_f32_bf16<<<(DM * DM / 4 + 255) / 256, 256, 0, stream>>>(Wk, wkb, DM * DM / 4);
  cvt_f32_bf16<<<(DM * DM / 4 + 255) / 256, 256, 0, stream>>>(Wv, wvb, DM * DM / 4);
  cvt_f32_bf16<<<(DM * DM / 4 + 255) / 256, 256, 0, stream>>>(Wo, wob, DM * DM / 4);

  // 2) QKV projections: 2048 waves (32-row tiles), 4 waves (128 thr) per block
  qkv_proj<<<512, 128, 0, stream>>>(xb, wqb, qb, 0);
  qkv_proj<<<512, 128, 0, stream>>>(xb, wkb, kb, 1);
  qkv_proj<<<512, 128, 0, stream>>>(xb, wvb, vtb, 2);

  // 3) flash attention: 2048 waves (32-row q tiles)
  attn_fwd<<<512, 128, 0, stream>>>(qb, kb, vtb, ctxb);

  // 4) output projection -> fp32 out
  o_proj<<<512, 128, 0, stream>>>(ctxb, wob, out);
}
